// SNNdecoder_1494648619482
// MI455X (gfx1250) — matrix operand order repaired and checked by reading, not compiled, not run on hardware
//
#include <hip/hip_runtime.h>
#include <hip/hip_bf16.h>

typedef __attribute__((ext_vector_type(16))) _Float16 v16h;
typedef __attribute__((ext_vector_type(8)))  _Float16 v8h;
typedef __attribute__((ext_vector_type(8)))  float    v8f;

#define TSTEPS 16
#define NBATCH 128

// ---------------------------------------------------------------------------
// B-operand packing: fragment-ready layout so each lane reads its 16 halves
// as one contiguous 32-byte chunk.
//   packed[((tileN*nk + kk)*32 + lane)*16 + i]  holds  B[k, n]  with
//   n = tileN*16 + (lane&15),  k = kk*32 + (lane>>4)*16 + i
// ---------------------------------------------------------------------------

// zz=[z;z] folded: value(k=c, n=o) = w_in[o,c] + w_in[o,c+128]; K=128 (nk=4)
__global__ void k_fold_w0(const float* __restrict__ w_in, _Float16* __restrict__ w0p) {
  int i = blockIdx.x * blockDim.x + threadIdx.x;        // 128*1024
  if (i >= 128 * 1024) return;
  int ii   = i & 15;
  int lane = (i >> 4) & 31;
  int kt   = i >> 9;                                    // tileN*4 + kk
  int kk   = kt & 3;
  int tileN = kt >> 2;
  int k = (kk << 5) + ((lane >> 4) << 4) + ii;          // = c
  int o = (tileN << 4) + (lane & 15);
  w0p[i] = (_Float16)(w_in[o * 256 + k] + w_in[o * 256 + k + 128]);
}

// conv weight (C_in, C_out, 3, 3): value(k = tap*C_in + ci, n = co)
__global__ void k_cast_wg(const float* __restrict__ w, _Float16* __restrict__ wg,
                          int C_in, int C_out) {
  int i = blockIdx.x * blockDim.x + threadIdx.x;
  int K = 9 * C_in;
  if (i >= K * C_out) return;
  int nkB  = K >> 5;
  int ii   = i & 15;
  int lane = (i >> 4) & 31;
  int kt   = i >> 9;
  int kk   = kt % nkB;
  int tileN = kt / nkB;
  int k  = (kk << 5) + ((lane >> 4) << 4) + ii;
  int co = (tileN << 4) + (lane & 15);
  int ci = k % C_in, tap = k / C_in;
  wg[i] = (_Float16)w[(ci * C_out + co) * 9 + tap];
}

// z (N,128,T) fp32 -> A0 [(n*T+t), c] f16 (row-major; A needs no packing)
__global__ void k_cast_z(const float* __restrict__ z, _Float16* __restrict__ zf) {
  int i = blockIdx.x * blockDim.x + threadIdx.x;        // 2048*128
  if (i >= NBATCH * TSTEPS * 128) return;
  int c = i & 127, nt = i >> 7;
  int n = nt >> 4, t = nt & 15;
  zf[(long)nt * 128 + c] = (_Float16)z[(n * 128 + c) * TSTEPS + t];
}

// ---------------------------------------------------------------------------
// WMMA GEMM, one wave32 per 16x32 output strip (two N-tiles share A).
//  conv_mode==0: dense C[M,N] = A[M,K] * B[K,N]
//  conv_mode==1: implicit ConvTranspose2d(k3,s2,p1,op1): rows are output
//    pixels; K tap-major. Each 32-wide K-step lies inside ONE tap (C_in is a
//    multiple of 32), so validity + row pointer are uniform per step and the
//    A fragment is two contiguous b128 loads (or zeros for invalid taps).
// ---------------------------------------------------------------------------
__global__ __launch_bounds__(256)
void k_gemm_wmma(const _Float16* __restrict__ A, const _Float16* __restrict__ Bp,
                 const float* __restrict__ bias, float* __restrict__ Cout,
                 int M, int K, int Nc, int conv_mode,
                 int C_in, int H_in, int W_in, int H_out, int W_out) {
  int lane = threadIdx.x & 31;
  int wave = blockIdx.x * (blockDim.x >> 5) + (threadIdx.x >> 5);
  int tilesN2 = Nc >> 5;                                 // N-tile pairs
  int totWork = (M >> 4) * tilesN2;
  if (wave >= totWork) return;                           // wave-uniform
  int tp    = wave % tilesN2;
  int tileM = wave / tilesN2;

  int half = lane >> 4;
  int lidx = lane & 15;
  int mrow = (tileM << 4) + lidx;
  int nk = K >> 5;

  const v8h* B8 = (const v8h*)Bp;
  long b0 = (long)(tp * 2) * nk * 64 + lane * 2;         // ((tileN0*nk+0)*32+lane)*2
  long b1 = b0 + (long)nk * 64;

  v8f acc0 = {}, acc1 = {};

  if (conv_mode == 0) {
    long abase = (long)mrow * K + (half << 4);
    for (int kk = 0; kk < nk; ++kk) {
      __builtin_prefetch(B8 + b0 + 64, 0, 1);
      v8h a0 = *(const v8h*)(A + abase + (kk << 5));
      v8h a1 = *(const v8h*)(A + abase + (kk << 5) + 8);
      v16h va = __builtin_shufflevector(a0, a1, 0,1,2,3,4,5,6,7,8,9,10,11,12,13,14,15);
      v8h p0 = B8[b0], p1 = B8[b0 + 1];
      v8h q0 = B8[b1], q1 = B8[b1 + 1];
      v16h vb0 = __builtin_shufflevector(p0, p1, 0,1,2,3,4,5,6,7,8,9,10,11,12,13,14,15);
      v16h vb1 = __builtin_shufflevector(q0, q1, 0,1,2,3,4,5,6,7,8,9,10,11,12,13,14,15);
      acc0 = __builtin_amdgcn_wmma_f32_16x16x32_f16(false, va, false, vb0, (short)0, acc0, false, false);
      acc1 = __builtin_amdgcn_wmma_f32_16x16x32_f16(false, va, false, vb1, (short)0, acc1, false, false);
      b0 += 64; b1 += 64;
    }
  } else {
    int P_out = H_out * W_out;
    int nt = mrow / P_out;
    int p  = mrow - nt * P_out;
    int oh = p / W_out;
    int ow = p - oh * W_out;
    int spt = C_in >> 5;                                 // K-steps per tap
    for (int tap = 0; tap < 9; ++tap) {
      int kh = tap / 3, kw = tap - kh * 3;
      int th = oh + 1 - kh, tw = ow + 1 - kw;
      int ih = th >> 1, iw = tw >> 1;
      bool valid = (((th | tw) >= 0) && !((th | tw) & 1) && ih < H_in && iw < W_in);
      const _Float16* ap = A + ((long)(nt * H_in + ih) * W_in + iw) * C_in + (half << 4);
      for (int s = 0; s < spt; ++s) {
        __builtin_prefetch(B8 + b0 + 64, 0, 1);
        v16h va = {};
        if (valid) {
          v8h a0 = *(const v8h*)(ap + (s << 5));
          v8h a1 = *(const v8h*)(ap + (s << 5) + 8);
          va = __builtin_shufflevector(a0, a1, 0,1,2,3,4,5,6,7,8,9,10,11,12,13,14,15);
        }
        v8h p0 = B8[b0], p1 = B8[b0 + 1];
        v8h q0 = B8[b1], q1 = B8[b1 + 1];
        v16h vb0 = __builtin_shufflevector(p0, p1, 0,1,2,3,4,5,6,7,8,9,10,11,12,13,14,15);
        v16h vb1 = __builtin_shufflevector(q0, q1, 0,1,2,3,4,5,6,7,8,9,10,11,12,13,14,15);
        acc0 = __builtin_amdgcn_wmma_f32_16x16x32_f16(false, va, false, vb0, (short)0, acc0, false, false);
        acc1 = __builtin_amdgcn_wmma_f32_16x16x32_f16(false, va, false, vb1, (short)0, acc1, false, false);
        b0 += 64; b1 += 64;
      }
    }
  }

  int coln0 = (tp << 5) + lidx;
  int coln1 = coln0 + 16;
  float bv0 = bias ? bias[coln0] : 0.f;
  float bv1 = bias ? bias[coln1] : 0.f;
  int rb = (tileM << 4) + (half << 3);                   // C layout: VGPR r -> M = rb + r
#pragma unroll
  for (int r = 0; r < 8; ++r) {
    Cout[(long)(rb + r) * Nc + coln0] = acc0[r] + bv0;
    Cout[(long)(rb + r) * Nc + coln1] = acc1[r] + bv1;
  }
}

// ---------------------------------------------------------------------------
// tdBN statistics: one block per channel; biased variance (matches jnp.var).
// ---------------------------------------------------------------------------
__global__ void k_bn_stats(const float* __restrict__ y, float* __restrict__ stats,
                           int rows, int C) {
  __shared__ float ss[256], sq[256];
  int c = blockIdx.x;
  float s = 0.f, q = 0.f;
  for (long r = threadIdx.x; r < rows; r += blockDim.x) {
    float v = y[r * C + c];
    s += v; q += v * v;
  }
  ss[threadIdx.x] = s; sq[threadIdx.x] = q;
  __syncthreads();
  for (int st = 128; st > 0; st >>= 1) {
    if ((int)threadIdx.x < st) {
      ss[threadIdx.x] += ss[threadIdx.x + st];
      sq[threadIdx.x] += sq[threadIdx.x + st];
    }
    __syncthreads();
  }
  if (threadIdx.x == 0) {
    float inv = 1.f / (float)rows;
    float m = ss[0] * inv;
    stats[2 * c]     = m;
    stats[2 * c + 1] = sq[0] * inv - m * m;
  }
}

// ---------------------------------------------------------------------------
// tdBN apply + LIF recurrence; emits f16 spikes in next layer's A layout.
// mode==1 fuses the 1024 -> (256,2,2) reshape of the linear layer output.
// ---------------------------------------------------------------------------
__global__ void k_bn_lif(const float* __restrict__ y, const float* __restrict__ stats,
                         const float* __restrict__ gamma, const float* __restrict__ beta,
                         _Float16* __restrict__ spikes, int P, int C, int mode) {
  long idx = (long)blockIdx.x * blockDim.x + threadIdx.x;
  long total = (long)NBATCH * P * C;
  if (idx >= total) return;
  int c = idx % C;
  long rem = idx / C;
  int p = rem % P;
  int n = rem / P;

  float mean = stats[2 * c], var = stats[2 * c + 1];
  float sc = 0.2f * rsqrtf(var + 1e-5f) * gamma[c];     // ALPHA*VTH = 0.2
  float bt = beta[c];

  float u = 0.f, o = 0.f;
  for (int t = 0; t < TSTEPS; ++t) {
    long row = (long)(n * TSTEPS + t) * P + p;
    float x = sc * (y[row * C + c] - mean) + bt;
    u = 0.25f * u * (1.f - o) + x;                      // TAU = 0.25
    o = (u > 0.2f) ? 1.f : 0.f;                         // VTH = 0.2
    if (mode == 0) {
      spikes[row * C + c] = (_Float16)o;
    } else {                                            // c = ci*4 + (ph*2+pw)
      int pp = c & 3, ci = c >> 2;
      long orow = (long)(n * TSTEPS + t) * 4 + pp;
      spikes[orow * 256 + ci] = (_Float16)o;
    }
  }
}

// ---------------------------------------------------------------------------
// Final ConvT(32->1, k3, s1, p1) + temporal weighted sum + tanh.
// ---------------------------------------------------------------------------
__global__ void k_final(const _Float16* __restrict__ sp, const float* __restrict__ w5,
                        const float* __restrict__ b5, float* __restrict__ out) {
  __shared__ float wsh[288];                            // (32,1,3,3)
  for (int i = threadIdx.x; i < 288; i += blockDim.x) wsh[i] = w5[i];
  __syncthreads();
  int idx = blockIdx.x * blockDim.x + threadIdx.x;      // 128*32*32 exact
  int ow = idx & 31, oh = (idx >> 5) & 31, n = idx >> 10;
  float bb = b5[0];
  float acc = 0.f, coef = 0.035184372088832f;           // 0.8^15
  for (int t = 0; t < TSTEPS; ++t) {
    int nt = n * TSTEPS + t;
    float yt = bb;
    for (int kh = 0; kh < 3; ++kh) {
      int ih = oh + 1 - kh;
      if ((unsigned)ih >= 32u) continue;
      for (int kw = 0; kw < 3; ++kw) {
        int iw = ow + 1 - kw;
        if ((unsigned)iw >= 32u) continue;
        int tap = kh * 3 + kw;
        const v8h* sr8 = (const v8h*)(sp + ((long)nt * 1024 + ih * 32 + iw) * 32);
#pragma unroll
        for (int ch = 0; ch < 4; ++ch) {
          v8h s = sr8[ch];
#pragma unroll
          for (int j = 0; j < 8; ++j)
            yt += (float)s[j] * wsh[(ch * 8 + j) * 9 + tap];
        }
      }
    }
    acc += yt * coef;
    coef *= 1.25f;                                      // advance 0.8^(15-t)
  }
  out[idx] = tanhf(acc);
}

// ---------------------------------------------------------------------------
// Workspace layout (bytes). Ping-pong:
//   P region (spikes f16): max A5 = 2048*1024*32*2  = 134217728
//   Q region (pre-act f32): max y4 = 2048*1024*32*4 = 268435456
// ---------------------------------------------------------------------------
static constexpr size_t OFF_W0    = 0;                       // 131072 f16
static constexpr size_t OFF_WG1   = OFF_W0  + 131072u * 2;
static constexpr size_t OFF_WG2   = OFF_WG1 + 294912u * 2;
static constexpr size_t OFF_WG3   = OFF_WG2 + 73728u  * 2;
static constexpr size_t OFF_WG4   = OFF_WG3 + 18432u  * 2;
static constexpr size_t OFF_STATS = OFF_WG4 + 9216u   * 2;
static constexpr size_t OFF_P     = 1064960;                 // 4K-aligned past stats
static constexpr size_t OFF_Q     = OFF_P + 134217728u;

extern "C" void kernel_launch(void* const* d_in, const int* in_sizes, int n_in,
                              void* d_out, int out_size, void* d_ws, size_t ws_size,
                              hipStream_t stream) {
  (void)in_sizes; (void)n_in; (void)out_size; (void)ws_size;
  const float* z     = (const float*)d_in[0];
  const float* w_in  = (const float*)d_in[2];
  const float* b_in  = (const float*)d_in[3];
  const float* g_in  = (const float*)d_in[4];
  const float* be_in = (const float*)d_in[5];
  const float* Wc[5] = { (const float*)d_in[6],  (const float*)d_in[10],
                         (const float*)d_in[14], (const float*)d_in[18],
                         (const float*)d_in[22] };
  const float* Bi[5] = { (const float*)d_in[7],  (const float*)d_in[11],
                         (const float*)d_in[15], (const float*)d_in[19],
                         (const float*)d_in[23] };
  const float* Ga[4] = { (const float*)d_in[8],  (const float*)d_in[12],
                         (const float*)d_in[16], (const float*)d_in[20] };
  const float* Be[4] = { (const float*)d_in[9],  (const float*)d_in[13],
                         (const float*)d_in[17], (const float*)d_in[21] };

  char* ws = (char*)d_ws;
  _Float16* w0p   = (_Float16*)(ws + OFF_W0);
  _Float16* wg[4] = { (_Float16*)(ws + OFF_WG1), (_Float16*)(ws + OFF_WG2),
                      (_Float16*)(ws + OFF_WG3), (_Float16*)(ws + OFF_WG4) };
  float*    stats = (float*)(ws + OFF_STATS);
  char*     bufP  = ws + OFF_P;     // f16 spikes / A operand
  char*     bufQ  = ws + OFF_Q;     // f32 pre-activations
  float*    outp  = (float*)d_out;

  // --- weight prep (fragment-ready packed B) ---
  k_fold_w0<<<(128 * 1024 + 255) / 256, 256, 0, stream>>>(w_in, w0p);
  const int CiL[4] = {256, 128, 64, 32};
  const int CoL[4] = {128, 64, 32, 32};
  const int HiL[4] = {2, 4, 8, 16};
  const int HoL[4] = {4, 8, 16, 32};
  for (int l = 0; l < 4; ++l) {
    int total = 9 * CiL[l] * CoL[l];
    k_cast_wg<<<(total + 255) / 256, 256, 0, stream>>>(Wc[l], wg[l], CiL[l], CoL[l]);
  }
  k_cast_z<<<(NBATCH * TSTEPS * 128 + 255) / 256, 256, 0, stream>>>(z, (_Float16*)bufP);

  // --- linear layer: (2048 x 128) x (128 x 1024) ---
  {
    int M = NBATCH * TSTEPS, K = 128, Nc = 1024;
    int work = (M >> 4) * (Nc >> 5);
    k_gemm_wmma<<<(work + 7) / 8, 256, 0, stream>>>(
        (const _Float16*)bufP, w0p, b_in, (float*)bufQ,
        M, K, Nc, /*conv=*/0, 0, 0, 0, 0, 0);
    k_bn_stats<<<1024, 256, 0, stream>>>((const float*)bufQ, stats, M, 1024);
    long neurons = (long)NBATCH * 1024;
    k_bn_lif<<<(int)((neurons + 255) / 256), 256, 0, stream>>>(
        (const float*)bufQ, stats, g_in, be_in, (_Float16*)bufP, 1, 1024, /*mode=*/1);
  }

  // --- four ConvT(s2) + tdBN + LIF stages ---
  for (int l = 0; l < 4; ++l) {
    int Ci = CiL[l], Co = CoL[l], Hi = HiL[l], Ho = HoL[l];
    int Pout = Ho * Ho;
    int M = NBATCH * TSTEPS * Pout;
    int K = 9 * Ci;
    int work = (M >> 4) * (Co >> 5);
    k_gemm_wmma<<<(work + 7) / 8, 256, 0, stream>>>(
        (const _Float16*)bufP, wg[l], Bi[l], (float*)bufQ,
        M, K, Co, /*conv=*/1, Ci, Hi, Hi, Ho, Ho);
    k_bn_stats<<<Co, 256, 0, stream>>>((const float*)bufQ, stats, M, Co);
    long neurons = (long)NBATCH * Pout * Co;
    k_bn_lif<<<(int)((neurons + 255) / 256), 256, 0, stream>>>(
        (const float*)bufQ, stats, Ga[l], Be[l], (_Float16*)bufP, Pout, Co, /*mode=*/0);
  }

  // --- final ConvT(s1) + temporal collapse + tanh ---
  k_final<<<(NBATCH * 1024) / 256, 256, 0, stream>>>(
      (const _Float16*)bufP, Wc[4], Bi[4], outp);
}
